// MultiheadAttention_9526237463071
// MI455X (gfx1250) — matrix operand order repaired and checked by reading, not compiled, not run on hardware
//
#include <hip/hip_runtime.h>

typedef __attribute__((ext_vector_type(16))) __bf16 v16bf;
typedef __attribute__((ext_vector_type(8)))  float  v8f;

#define D_MODEL 128
#define SEQ     4096
#define ROWS    8192      // B*S
#define HDIM    32
#define NHEADPAIR 8       // B*NUM_HEAD contiguous head panels

// ---------------------------------------------------------------------------
// Embedding gather + sinusoidal positional encoding -> bf16 x [8192][128]
// ---------------------------------------------------------------------------
__global__ __launch_bounds__(256) void embed_pe_kernel(
    const int* __restrict__ seq, const float* __restrict__ emb,
    unsigned short* __restrict__ xb_u) {
  __bf16* xb = reinterpret_cast<__bf16*>(xb_u);
  int g = blockIdx.x * blockDim.x + threadIdx.x;   // 0 .. ROWS*D_MODEL-1
  int d = g & (D_MODEL - 1);
  int row = g >> 7;                                 // b*SEQ + s
  int s = row & (SEQ - 1);
  int tok = seq[row];
  float val = emb[(size_t)tok * D_MODEL + d];
  // pe[s, d]: angle = s / 10000^((d&~1)/128); even->sin, odd->cos
  float freq = __powf(10000.0f, -(float)(d & ~1) * (1.0f / 128.0f));
  float ang = (float)s * freq;
  val += (d & 1) ? __cosf(ang) : __sinf(ang);
  xb[g] = (__bf16)val;
}

// ---------------------------------------------------------------------------
// fp32 -> bf16 weight conversion
// ---------------------------------------------------------------------------
__global__ __launch_bounds__(256) void cvt_bf16_kernel(
    const float* __restrict__ a, unsigned short* __restrict__ b_u, int n) {
  __bf16* b = reinterpret_cast<__bf16*>(b_u);
  int g = blockIdx.x * blockDim.x + threadIdx.x;
  if (g < n) b[g] = (__bf16)a[g];
}

// ---------------------------------------------------------------------------
// y = x @ W^T   (x: [8192][128] bf16, W: [128][128] bf16, torch layout W[e][d])
// One block = 16-row M tile, 4 waves; wave w owns output columns [32w, 32w+32).
// A fragments (4 K-steps of the X row) hoisted and shared by both 16-col tiles.
// Output type is a template parameter -> branch-free store clauses.
// ---------------------------------------------------------------------------
template <bool F32OUT>
__global__ __launch_bounds__(128) void proj_kernel(
    const unsigned short* __restrict__ Xu, const unsigned short* __restrict__ Wu,
    unsigned short* __restrict__ Ybu, float* __restrict__ Yf) {
  const __bf16* X = reinterpret_cast<const __bf16*>(Xu);
  const __bf16* W = reinterpret_cast<const __bf16*>(Wu);
  __bf16* Yb = reinterpret_cast<__bf16*>(Ybu);

  const int tid  = threadIdx.x;
  const int wave = tid >> 5;
  const int lane = tid & 31;
  const int half = lane >> 4;
  const int ln   = lane & 15;
  const int mbase = blockIdx.x * 16;

  // A fragments for all 4 K-steps (each: row M = ln, 16-bit A layout)
  v16bf a[4];
  #pragma unroll
  for (int kb2 = 0; kb2 < 4; ++kb2) {
    #pragma unroll
    for (int i = 0; i < 16; ++i) {
      int kk = (i < 8) ? (8 * half + i) : (16 + 8 * half + (i - 8));
      a[kb2][i] = X[(size_t)(mbase + ln) * D_MODEL + kb2 * 32 + kk];
    }
  }

  #pragma unroll
  for (int which = 0; which < 2; ++which) {
    const int nbase = wave * 32 + which * 16;
    v8f c = {};
    #pragma unroll
    for (int kb2 = 0; kb2 < 4; ++kb2) {
      v16bf b;
      #pragma unroll
      for (int i = 0; i < 16; ++i) {
        // B fragment: B(k=d, n=e) = W[e][d]; col n = ln, K in table order
        int kd = (i < 8) ? (8 * half + i) : (16 + 8 * half + (i - 8));
        b[i] = W[(size_t)(nbase + ln) * D_MODEL + kb2 * 32 + kd];
      }
      c = __builtin_amdgcn_wmma_f32_16x16x32_bf16(false, a[kb2], false, b,
                                                  (short)0, c, false, false);
    }
    #pragma unroll
    for (int r = 0; r < 8; ++r) {
      int row = mbase + r + 8 * half;
      int col = nbase + ln;
      if constexpr (F32OUT) Yf[(size_t)row * D_MODEL + col] = c[r];
      else                  Yb[(size_t)row * D_MODEL + col] = (__bf16)c[r];
    }
  }
}

// ---------------------------------------------------------------------------
// Flash attention per head panel. Q/K/V/O head panels are contiguous
// [4096][32] row-major slices (the reference's "plain view" reshape).
// Grid: (64 q-blocks, 8 head panels), 128 threads = 4 waves x 16 q rows.
// masking: raw = |s|*scale, -inf above diagonal; online softmax.
// ---------------------------------------------------------------------------
__global__ __launch_bounds__(128) void attn_kernel(
    const unsigned short* __restrict__ Qu, const unsigned short* __restrict__ Ku,
    const unsigned short* __restrict__ Vu, unsigned short* __restrict__ Ou) {
  const int head = blockIdx.y;                 // b*4 + h
  const int qbase = blockIdx.x * 64;
  const size_t hoff = (size_t)head * SEQ * HDIM;
  const __bf16* Qh = reinterpret_cast<const __bf16*>(Qu) + hoff;
  const __bf16* Kh = reinterpret_cast<const __bf16*>(Ku) + hoff;
  const __bf16* Vh = reinterpret_cast<const __bf16*>(Vu) + hoff;
  __bf16*       Oh = reinterpret_cast<__bf16*>(Ou) + hoff;

  const int tid  = threadIdx.x;
  const int wave = tid >> 5;
  const int lane = tid & 31;
  const int half = lane >> 4;
  const int ln   = lane & 15;

  __shared__ __bf16 ksh[32 * 32];        // K tile, row-major [key][d]
  __shared__ __bf16 vsh[32 * 32];        // V tile transposed [d][key]
  __shared__ __bf16 psh[4][16 * 32];     // per-wave P staging (C->A relayout)

  const int wq   = qbase + wave * 16;    // wave's first query row
  const int qrow = wq + ln;              // this lane's A-fragment row

  // Q A-fragment, loaded once (rows are 32 contiguous bf16 in the head panel)
  v16bf afq;
  #pragma unroll
  for (int i = 0; i < 16; ++i) {
    int kk = (i < 8) ? (8 * half + i) : (16 + 8 * half + (i - 8));
    afq[i] = Qh[(size_t)qrow * HDIM + kk];
  }

  float mstat[8], lstat[8];
  v8f o0 = {}, o1 = {};
  #pragma unroll
  for (int r = 0; r < 8; ++r) { mstat[r] = -__builtin_inff(); lstat[r] = 0.0f; }

  const float scale = 0.17677669529663687f;   // 1/sqrt(32)
  const int nkb = (qbase + 64) / 32;          // causal key-block count

  for (int kbk = 0; kbk < nkb; ++kbk) {
    __syncthreads();   // protect previous iteration's LDS reads
    {
      const __bf16* kg = Kh + (size_t)kbk * 32 * HDIM;
      const __bf16* vg = Vh + (size_t)kbk * 32 * HDIM;
      #pragma unroll
      for (int j = 0; j < 8; ++j) {            // 2KB straight copy (tile rows
        int idx = tid * 8 + j;                 // are contiguous in the panel)
        ksh[idx] = kg[idx];
      }
      #pragma unroll
      for (int j = 0; j < 8; ++j) {            // transpose V into LDS
        int idx = tid * 8 + j;
        int r = idx >> 5, c = idx & 31;
        vsh[c * 32 + r] = vg[idx];
      }
      if (kbk + 1 < 128) {                     // prefetch next K/V block
        __builtin_prefetch(kg + 1024, 0, 1);
        __builtin_prefetch(vg + 1024, 0, 1);
      }
    }
    __syncthreads();

    // wave-uniform causal skip (keeps EXEC all-ones around WMMA)
    if (32 * kbk > wq + 15) continue;

    // scores: S[16q x 32k] = Q . K^T  (two N=16 WMMAs)
    v16bf bk0, bk1;
    #pragma unroll
    for (int i = 0; i < 16; ++i) {
      int kd = (i < 8) ? (8 * half + i) : (16 + 8 * half + (i - 8));                  // dim index 0..31
      bk0[i] = ksh[ln * 32 + kd];              // key = ln
      bk1[i] = ksh[(16 + ln) * 32 + kd];       // key = 16 + ln
    }
    v8f s0 = {}, s1 = {};
    s0 = __builtin_amdgcn_wmma_f32_16x16x32_bf16(false, afq, false, bk0,
                                                 (short)0, s0, false, false);
    s1 = __builtin_amdgcn_wmma_f32_16x16x32_bf16(false, afq, false, bk1,
                                                 (short)0, s1, false, false);

    // abs * scale + causal mask; rowwise online softmax
    float p0[8], p1[8], rmax[8];
    #pragma unroll
    for (int r = 0; r < 8; ++r) {
      int m = wq + r + 8 * half;               // global query row
      int k0 = kbk * 32 + ln;
      int k1 = kbk * 32 + 16 + ln;
      float v0 = fabsf(s0[r]) * scale;
      float v1 = fabsf(s1[r]) * scale;
      v0 = (k0 <= m) ? v0 : -__builtin_inff();
      v1 = (k1 <= m) ? v1 : -__builtin_inff();
      p0[r] = v0; p1[r] = v1;
      rmax[r] = fmaxf(v0, v1);
    }
    #pragma unroll
    for (int off = 1; off < 16; off <<= 1) {   // reduce across 16-lane half
      #pragma unroll
      for (int r = 0; r < 8; ++r)
        rmax[r] = fmaxf(rmax[r], __shfl_xor(rmax[r], off, 32));
    }
    float alpha[8], rsum[8];
    #pragma unroll
    for (int r = 0; r < 8; ++r) {
      float mn = fmaxf(mstat[r], rmax[r]);
      alpha[r] = __expf(mstat[r] - mn);
      mstat[r] = mn;
      p0[r] = __expf(p0[r] - mn);
      p1[r] = __expf(p1[r] - mn);
      rsum[r] = p0[r] + p1[r];
    }
    #pragma unroll
    for (int off = 1; off < 16; off <<= 1) {
      #pragma unroll
      for (int r = 0; r < 8; ++r)
        rsum[r] += __shfl_xor(rsum[r], off, 32);
    }
    #pragma unroll
    for (int r = 0; r < 8; ++r) {
      lstat[r] = lstat[r] * alpha[r] + rsum[r];
      o0[r] = o0[r] * alpha[r];
      o1[r] = o1[r] * alpha[r];
    }

    // C-layout -> A-layout for P via per-wave LDS tile
    __bf16* pw = psh[wave];
    #pragma unroll
    for (int r = 0; r < 8; ++r) {
      int m = r + 8 * half;
      pw[m * 32 + ln]      = (__bf16)p0[r];
      pw[m * 32 + 16 + ln] = (__bf16)p1[r];
    }
    __builtin_amdgcn_wave_barrier();           // intra-wave DS ordering fence
    v16bf ap, bv0, bv1;
    #pragma unroll
    for (int i = 0; i < 16; ++i) {
      int kk = (i < 8) ? (8 * half + i) : (16 + 8 * half + (i - 8));
      ap[i] = pw[ln * 32 + kk];                // A row = ln, K = key index
      int kv = kk;                  // key index
      bv0[i] = vsh[ln * 32 + kv];              // d = ln
      bv1[i] = vsh[(16 + ln) * 32 + kv];       // d = 16 + ln
    }
    o0 = __builtin_amdgcn_wmma_f32_16x16x32_bf16(false, ap, false, bv0,
                                                 (short)0, o0, false, false);
    o1 = __builtin_amdgcn_wmma_f32_16x16x32_bf16(false, ap, false, bv1,
                                                 (short)0, o1, false, false);
  }

  // finalize: O = acc / l, store bf16 into the contiguous head panel
  #pragma unroll
  for (int r = 0; r < 8; ++r) {
    int m = wq + r + 8 * half;
    float inv = 1.0f / lstat[r];
    Oh[(size_t)m * HDIM + ln]      = (__bf16)(o0[r] * inv);
    Oh[(size_t)m * HDIM + 16 + ln] = (__bf16)(o1[r] * inv);
  }
}

// ---------------------------------------------------------------------------
// Host launcher
// ---------------------------------------------------------------------------
extern "C" void kernel_launch(void* const* d_in, const int* in_sizes, int n_in,
                              void* d_out, int out_size, void* d_ws, size_t ws_size,
                              hipStream_t stream) {
  (void)in_sizes; (void)n_in; (void)out_size; (void)ws_size;
  const int*   seq = (const int*)d_in[0];
  const float* emb = (const float*)d_in[1];
  const float* wq  = (const float*)d_in[2];
  const float* wk  = (const float*)d_in[3];
  const float* wv  = (const float*)d_in[4];
  const float* wo  = (const float*)d_in[5];
  float* out = (float*)d_out;

  char* p = (char*)d_ws;
  const size_t act_bytes = (size_t)ROWS * D_MODEL * sizeof(unsigned short);
  const size_t w_bytes   = (size_t)D_MODEL * D_MODEL * sizeof(unsigned short);
  unsigned short* xb  = (unsigned short*)p; p += act_bytes;
  unsigned short* qb  = (unsigned short*)p; p += act_bytes;
  unsigned short* kb  = (unsigned short*)p; p += act_bytes;
  unsigned short* vb  = (unsigned short*)p; p += act_bytes;
  unsigned short* ob  = (unsigned short*)p; p += act_bytes;
  unsigned short* wqb = (unsigned short*)p; p += w_bytes;
  unsigned short* wkb = (unsigned short*)p; p += w_bytes;
  unsigned short* wvb = (unsigned short*)p; p += w_bytes;
  unsigned short* wob = (unsigned short*)p; p += w_bytes;

  const int wn = D_MODEL * D_MODEL;
  cvt_bf16_kernel<<<(wn + 255) / 256, 256, 0, stream>>>(wq, wqb, wn);
  cvt_bf16_kernel<<<(wn + 255) / 256, 256, 0, stream>>>(wk, wkb, wn);
  cvt_bf16_kernel<<<(wn + 255) / 256, 256, 0, stream>>>(wv, wvb, wn);
  cvt_bf16_kernel<<<(wn + 255) / 256, 256, 0, stream>>>(wo, wob, wn);

  embed_pe_kernel<<<(ROWS * D_MODEL) / 256, 256, 0, stream>>>(seq, emb, xb);

  proj_kernel<false><<<ROWS / 16, 128, 0, stream>>>(xb, wqb, qb, nullptr);
  proj_kernel<false><<<ROWS / 16, 128, 0, stream>>>(xb, wkb, kb, nullptr);
  proj_kernel<false><<<ROWS / 16, 128, 0, stream>>>(xb, wvb, vb, nullptr);

  attn_kernel<<<dim3(SEQ / 64, NHEADPAIR), 128, 0, stream>>>(qb, kb, vb, ob);

  proj_kernel<true><<<ROWS / 16, 128, 0, stream>>>(ob, wob, nullptr, out);
}
